// CausalSelfAttention_67748814127162
// MI455X (gfx1250) — matrix operand order repaired and checked by reading, not compiled, not run on hardware
//
#include <hip/hip_runtime.h>
#include <hip/hip_bf16.h>

typedef __attribute__((ext_vector_type(16))) _Float16 v16h;
typedef __attribute__((ext_vector_type(8)))  float    v8f;
typedef unsigned int u32;
typedef __attribute__((ext_vector_type(4))) u32 v4u;
typedef __attribute__((ext_vector_type(8))) int v8i;
typedef __attribute__((ext_vector_type(4))) int v4i;

#define DEVINL __device__ __forceinline__

// ---- constants for this problem ----
#define BATCH 4
#define SEQ   2048
#define CDIM  1024
#define NHEAD 16
#define HDIM  64
#define MROWS (BATCH * SEQ)      // 8192
#define N3    (3 * CDIM)         // 3072

// ------------------------------------------------------------------
// Fragment loaders per CDNA5 WMMA VGPR layouts (16-bit, 16x16x32).
// A (16x32, MxK): lane l<16 -> row M=l, K {k0..k0+7, k0+16..k0+23};
//                 lane l>=16 -> same rows, K {+8, +24}.
// B (32x16, KxN): lane l -> column N=l%16; lanes<16 hold K {0..7, 16..23},
//                 lanes>=16 hold K {8..15, 24..31} (same as A).
// C/D (16x16 f32): lane l -> col N=l%16; VGPR v -> row M=v+8*(l/16).
// ------------------------------------------------------------------
DEVINL v16h load_frag_a(const _Float16* rowk0, int hi) {
  const _Float16* p = rowk0 + (hi ? 8 : 0);
  v16h f;
  reinterpret_cast<uint4*>(&f)[0] = *reinterpret_cast<const uint4*>(p);
  reinterpret_cast<uint4*>(&f)[1] = *reinterpret_cast<const uint4*>(p + 16);
  return f;
}

DEVINL v16h load_frag_b(const _Float16* colk0, int hi) {
  const _Float16* p = colk0 + (hi ? 8 : 0);
  v16h f;
  reinterpret_cast<uint4*>(&f)[0] = *reinterpret_cast<const uint4*>(p);
  reinterpret_cast<uint4*>(&f)[1] = *reinterpret_cast<const uint4*>(p + 16);
  return f;
}

DEVINL v8f wmma_f16(v16h a, v16h b, v8f c) {
  return __builtin_amdgcn_wmma_f32_16x16x32_f16(
      false, a, false, b, (short)0, c, false, false);
}

// ------------------------------------------------------------------
// Tensor Data Mover: 2-D tile load (global f16 tensor -> LDS tile).
// D# per CDNA5 ISA:
//   group0: count=1 | lds_addr | global_addr[56:0] | type=2
//   group1: data_size=1 (2B) | tensor_dim0/1 | tile_dim0/1 | dim0 stride
//   groups 2/3: zero (2-D tensor)
// 6-arg builtin variant (clang-23 / therock-10.0 headers).
// ------------------------------------------------------------------
DEVINL void tdm_load_2d(u32 lds_off, const _Float16* gptr,
                        u32 tensor_d0, u32 tensor_d1,
                        u32 tile_d0, u32 tile_d1, u32 stride0) {
  unsigned long long ga = (unsigned long long)(uintptr_t)gptr;
  v4u g0;
  g0[0] = 1u;                                            // count=1 (valid), user mode
  g0[1] = lds_off;                                       // lds_addr (bytes)
  g0[2] = (u32)(ga & 0xffffffffu);                       // global_addr[31:0]
  g0[3] = (u32)((ga >> 32) & 0x01ffffffu) | (2u << 30);  // addr[56:32] | type=2
  v8i g1;
  g1[0] = (int)(1u << 16);                               // data_size=1 -> 2 bytes
  g1[1] = (int)(tensor_d0 << 16);                        // tensor_dim0[15:0]
  g1[2] = (int)((tensor_d0 >> 16) | (tensor_d1 << 16));  // dim0 hi | dim1 lo
  g1[3] = (int)((tensor_d1 >> 16) | (tile_d0 << 16));    // dim1 hi | tile_dim0
  g1[4] = (int)(tile_d1 & 0xffffu);                      // tile_dim1 (tile_dim2=0)
  g1[5] = (int)stride0;                                  // tensor_dim0_stride[31:0]
  g1[6] = 0;                                             // stride0 hi | stride1 lo
  g1[7] = 0;                                             // stride1 hi
  v4i z4 = {};
  v8i z8 = {};
  __builtin_amdgcn_tensor_load_to_lds(g0, g1, z4, z4, z8, 0);
}

// ------------------------------------------------------------------
// fp32 -> f16 conversion
// ------------------------------------------------------------------
__global__ __launch_bounds__(256) void cvt_f32_to_f16(
    const float* __restrict__ src, _Float16* __restrict__ dst, int n) {
  int i = blockIdx.x * 256 + threadIdx.x;
  if (i < n) dst[i] = (_Float16)src[i];
}

// ------------------------------------------------------------------
// Fused QKV projection GEMM: out[m, o] = sum_c x[m,c] * Wcat[o,c] + b[o]
// M = 8192, N = 3072 (q|k|v), K = 1024.
// Block tile 128x128, 8 waves, wave tile 64x32. Both 128x32 input
// tiles staged into LDS by the TDM (double-buffered), fragments read
// with ds_load_b128.
// ------------------------------------------------------------------
__global__ __launch_bounds__(256) void qkv_gemm_kernel(
    const _Float16* __restrict__ x16, const _Float16* __restrict__ Wcat,
    const float* __restrict__ bq, const float* __restrict__ bk,
    const float* __restrict__ bv,
    _Float16* __restrict__ q16, _Float16* __restrict__ k16,
    _Float16* __restrict__ vT16,
    float* __restrict__ presentK, float* __restrict__ presentV) {
  __shared__ __align__(128) _Float16 xs[2][128 * 32];
  __shared__ __align__(128) _Float16 wsb[2][128 * 32];

  const int lane = threadIdx.x & 31;
  const int wave = threadIdx.x >> 5;
  const int hi   = lane >> 4;
  const int ln   = lane & 15;
  const int MbaseB = blockIdx.y * 128;
  const int NbaseB = blockIdx.x * 128;
  const int mw = (wave >> 2) * 64;   // wave row offset inside 128-row tile
  const int nw = (wave & 3) * 32;    // wave col offset inside 128-col tile

  const u32 xs0 = (u32)(uintptr_t)&xs[0][0];
  const u32 xs1 = (u32)(uintptr_t)&xs[1][0];
  const u32 ws0 = (u32)(uintptr_t)&wsb[0][0];
  const u32 ws1 = (u32)(uintptr_t)&wsb[1][0];

  const v8f z = {};
  v8f acc[4][2];
#pragma unroll
  for (int sm = 0; sm < 4; ++sm)
#pragma unroll
    for (int sn = 0; sn < 2; ++sn) acc[sm][sn] = z;

  // prime buffer 0
  if (wave == 0) {
    tdm_load_2d(xs0, x16 + (size_t)MbaseB * CDIM, CDIM, MROWS, 32, 128, CDIM);
    tdm_load_2d(ws0, Wcat + (size_t)NbaseB * CDIM, CDIM, N3, 32, 128, CDIM);
  }

  int buf = 0;
  for (int k0 = 0; k0 < CDIM; k0 += 32, buf ^= 1) {
    if (wave == 0) {
      if (k0 + 32 < CDIM) {
        // overlap: issue next tile pair into the other buffer, then wait
        // only for the older pair (TDM completes in order per wave).
        tdm_load_2d(buf ? xs0 : xs1,
                    x16 + (size_t)MbaseB * CDIM + (k0 + 32), CDIM, MROWS, 32, 128, CDIM);
        tdm_load_2d(buf ? ws0 : ws1,
                    Wcat + (size_t)NbaseB * CDIM + (k0 + 32), CDIM, N3, 32, 128, CDIM);
        __builtin_amdgcn_s_wait_tensorcnt(2);
      } else {
        __builtin_amdgcn_s_wait_tensorcnt(0);
      }
    }
    __syncthreads();   // publish LDS tiles to all waves

    v16h a[4], b[2];
#pragma unroll
    for (int sm = 0; sm < 4; ++sm)
      a[sm] = load_frag_a(&xs[buf][(mw + sm * 16 + ln) * 32], hi);
#pragma unroll
    for (int sn = 0; sn < 2; ++sn)
      b[sn] = load_frag_b(&wsb[buf][(nw + sn * 16 + ln) * 32], hi);
#pragma unroll
    for (int sm = 0; sm < 4; ++sm)
#pragma unroll
      for (int sn = 0; sn < 2; ++sn)
        acc[sm][sn] = wmma_f16(a[sm], b[sn], acc[sm][sn]);

    __syncthreads();   // everyone done reading before this buffer is refilled
  }

  // Epilogue: scatter into head-major q/k (and transposed v), write present.
#pragma unroll
  for (int sm = 0; sm < 4; ++sm) {
#pragma unroll
    for (int sn = 0; sn < 2; ++sn) {
      const int ng = NbaseB + nw + sn * 16 + ln;
      float bias;
      if (ng < CDIM)            bias = bq[ng];
      else if (ng < 2 * CDIM)   bias = bk[ng - CDIM];
      else                      bias = bv[ng - 2 * CDIM];
#pragma unroll
      for (int v = 0; v < 8; ++v) {
        const int mg = MbaseB + mw + sm * 16 + v + 8 * hi;
        const int bb = mg >> 11;           // mg / SEQ
        const int t  = mg & (SEQ - 1);
        const float val = acc[sm][sn][v] + bias;
        if (ng < CDIM) {
          const int h = ng >> 6, d = ng & 63;
          q16[(((size_t)(bb * NHEAD + h)) * SEQ + t) * HDIM + d] = (_Float16)val;
        } else if (ng < 2 * CDIM) {
          const int o = ng - CDIM;
          const int h = o >> 6, d = o & 63;
          const size_t idx = (((size_t)(bb * NHEAD + h)) * SEQ + t) * HDIM + d;
          k16[idx] = (_Float16)val;
          presentK[idx] = val;
        } else {
          const int o = ng - 2 * CDIM;
          const int h = o >> 6, d = o & 63;
          const size_t idx = (((size_t)(bb * NHEAD + h)) * SEQ + t) * HDIM + d;
          vT16[(((size_t)(bb * NHEAD + h)) * HDIM + d) * SEQ + t] = (_Float16)val;
          presentV[idx] = val;
        }
      }
    }
  }
}

// ------------------------------------------------------------------
// Flash attention: per (b,h), each wave owns 16 query rows.
// S = q k^T (2 WMMAs per 16-key subtile over D=64), online softmax with
// shfl_xor row reductions, P staged through LDS (C-layout -> A-layout),
// O accumulated with 4 WMMAs per 32-key step.
// ------------------------------------------------------------------
__global__ __launch_bounds__(256) void attn_kernel(
    const _Float16* __restrict__ q16, const _Float16* __restrict__ k16,
    const _Float16* __restrict__ vT16, _Float16* __restrict__ y16) {
  __shared__ __align__(16) _Float16 plds[8][16 * 32];

  const int lane = threadIdx.x & 31;
  const int wave = threadIdx.x >> 5;
  const int hi   = lane >> 4;
  const int ln   = lane & 15;
  const int bh = blockIdx.y;
  const int b  = bh >> 4;
  const int h  = bh & 15;
  const int q0 = blockIdx.x * 128 + wave * 16;

  const _Float16* qp = q16 + (size_t)bh * SEQ * HDIM;
  const _Float16* kp = k16 + (size_t)bh * SEQ * HDIM;
  const _Float16* vp = vT16 + (size_t)bh * HDIM * SEQ;

  const v16h aq0 = load_frag_a(qp + (size_t)(q0 + ln) * HDIM + 0, hi);
  const v16h aq1 = load_frag_a(qp + (size_t)(q0 + ln) * HDIM + 32, hi);

  const v8f z = {};
  v8f o0 = z, o1 = z, o2 = z, o3 = z;
  float mrun[8], lrun[8];
#pragma unroll
  for (int v = 0; v < 8; ++v) { mrun[v] = -1.0e30f; lrun[v] = 0.0f; }

  const float scale = 0.125f;   // 1/sqrt(64)
  const int kend = q0 + 16;

  for (int kb = 0; kb < kend; kb += 32) {
    v8f s0 = z, s1 = z;
    s0 = wmma_f16(aq0, load_frag_b(kp + (size_t)(kb + ln) * HDIM + 0, hi), s0);
    s0 = wmma_f16(aq1, load_frag_b(kp + (size_t)(kb + ln) * HDIM + 32, hi), s0);
    s1 = wmma_f16(aq0, load_frag_b(kp + (size_t)(kb + 16 + ln) * HDIM + 0, hi), s1);
    s1 = wmma_f16(aq1, load_frag_b(kp + (size_t)(kb + 16 + ln) * HDIM + 32, hi), s1);

#pragma unroll
    for (int v = 0; v < 8; ++v) {
      const int qi = q0 + v + 8 * hi;
      float e0 = s0[v] * scale; if (kb + ln > qi)      e0 = -1.0e30f;
      float e1 = s1[v] * scale; if (kb + 16 + ln > qi) e1 = -1.0e30f;
      float rm = fmaxf(e0, e1);
      rm = fmaxf(rm, __shfl_xor(rm, 1, 32));
      rm = fmaxf(rm, __shfl_xor(rm, 2, 32));
      rm = fmaxf(rm, __shfl_xor(rm, 4, 32));
      rm = fmaxf(rm, __shfl_xor(rm, 8, 32));
      const float mnew = fmaxf(mrun[v], rm);
      const float corr = __expf(mrun[v] - mnew);
      mrun[v] = mnew;
      const float p0 = __expf(e0 - mnew);
      const float p1 = __expf(e1 - mnew);
      float rs = p0 + p1;
      rs += __shfl_xor(rs, 1, 32);
      rs += __shfl_xor(rs, 2, 32);
      rs += __shfl_xor(rs, 4, 32);
      rs += __shfl_xor(rs, 8, 32);
      lrun[v] = lrun[v] * corr + rs;
      o0[v] *= corr; o1[v] *= corr; o2[v] *= corr; o3[v] *= corr;
      const int m = v + 8 * hi;
      plds[wave][m * 32 + ln]      = (_Float16)p0;
      plds[wave][m * 32 + 16 + ln] = (_Float16)p1;
    }

    const v16h pa = load_frag_a(&plds[wave][ln * 32], hi);

    o0 = wmma_f16(pa, load_frag_b(vp + (size_t)(0  + ln) * SEQ + kb, hi), o0);
    o1 = wmma_f16(pa, load_frag_b(vp + (size_t)(16 + ln) * SEQ + kb, hi), o1);
    o2 = wmma_f16(pa, load_frag_b(vp + (size_t)(32 + ln) * SEQ + kb, hi), o2);
    o3 = wmma_f16(pa, load_frag_b(vp + (size_t)(48 + ln) * SEQ + kb, hi), o3);
  }

#pragma unroll
  for (int v = 0; v < 8; ++v) {
    const float inv = 1.0f / lrun[v];
    const int t = q0 + v + 8 * hi;
    const size_t row = ((size_t)b * SEQ + t) * CDIM + h * HDIM;
    y16[row + 0  + ln] = (_Float16)(o0[v] * inv);
    y16[row + 16 + ln] = (_Float16)(o1[v] * inv);
    y16[row + 32 + ln] = (_Float16)(o2[v] * inv);
    y16[row + 48 + ln] = (_Float16)(o3[v] * inv);
  }
}

// ------------------------------------------------------------------
// Output projection (TDM-staged like the QKV GEMM): fp32 out + bias.
// ------------------------------------------------------------------
__global__ __launch_bounds__(256) void proj_gemm_kernel(
    const _Float16* __restrict__ y16, const _Float16* __restrict__ Wp16,
    const float* __restrict__ bp, float* __restrict__ out) {
  __shared__ __align__(128) _Float16 xs[2][128 * 32];
  __shared__ __align__(128) _Float16 wsb[2][128 * 32];

  const int lane = threadIdx.x & 31;
  const int wave = threadIdx.x >> 5;
  const int hi   = lane >> 4;
  const int ln   = lane & 15;
  const int MbaseB = blockIdx.y * 128;
  const int NbaseB = blockIdx.x * 128;
  const int mw = (wave >> 2) * 64;
  const int nw = (wave & 3) * 32;

  const u32 xs0 = (u32)(uintptr_t)&xs[0][0];
  const u32 xs1 = (u32)(uintptr_t)&xs[1][0];
  const u32 ws0 = (u32)(uintptr_t)&wsb[0][0];
  const u32 ws1 = (u32)(uintptr_t)&wsb[1][0];

  const v8f z = {};
  v8f acc[4][2];
#pragma unroll
  for (int sm = 0; sm < 4; ++sm)
#pragma unroll
    for (int sn = 0; sn < 2; ++sn) acc[sm][sn] = z;

  if (wave == 0) {
    tdm_load_2d(xs0, y16 + (size_t)MbaseB * CDIM, CDIM, MROWS, 32, 128, CDIM);
    tdm_load_2d(ws0, Wp16 + (size_t)NbaseB * CDIM, CDIM, CDIM, 32, 128, CDIM);
  }

  int buf = 0;
  for (int k0 = 0; k0 < CDIM; k0 += 32, buf ^= 1) {
    if (wave == 0) {
      if (k0 + 32 < CDIM) {
        tdm_load_2d(buf ? xs0 : xs1,
                    y16 + (size_t)MbaseB * CDIM + (k0 + 32), CDIM, MROWS, 32, 128, CDIM);
        tdm_load_2d(buf ? ws0 : ws1,
                    Wp16 + (size_t)NbaseB * CDIM + (k0 + 32), CDIM, CDIM, 32, 128, CDIM);
        __builtin_amdgcn_s_wait_tensorcnt(2);
      } else {
        __builtin_amdgcn_s_wait_tensorcnt(0);
      }
    }
    __syncthreads();

    v16h a[4], b[2];
#pragma unroll
    for (int sm = 0; sm < 4; ++sm)
      a[sm] = load_frag_a(&xs[buf][(mw + sm * 16 + ln) * 32], hi);
#pragma unroll
    for (int sn = 0; sn < 2; ++sn)
      b[sn] = load_frag_b(&wsb[buf][(nw + sn * 16 + ln) * 32], hi);
#pragma unroll
    for (int sm = 0; sm < 4; ++sm)
#pragma unroll
      for (int sn = 0; sn < 2; ++sn)
        acc[sm][sn] = wmma_f16(a[sm], b[sn], acc[sm][sn]);

    __syncthreads();
  }

#pragma unroll
  for (int sm = 0; sm < 4; ++sm) {
#pragma unroll
    for (int sn = 0; sn < 2; ++sn) {
      const int ng = NbaseB + nw + sn * 16 + ln;
      const float bias = bp[ng];
#pragma unroll
      for (int v = 0; v < 8; ++v) {
        const int mg = MbaseB + mw + sm * 16 + v + 8 * hi;
        out[(size_t)mg * CDIM + ng] = acc[sm][sn][v] + bias;
      }
    }
  }
}

// ------------------------------------------------------------------
extern "C" void kernel_launch(void* const* d_in, const int* in_sizes, int n_in,
                              void* d_out, int out_size, void* d_ws, size_t ws_size,
                              hipStream_t stream) {
  (void)in_sizes; (void)n_in; (void)out_size; (void)ws_size;

  const float* x  = (const float*)d_in[0];
  const float* Wq = (const float*)d_in[1];
  const float* bq = (const float*)d_in[2];
  const float* Wk = (const float*)d_in[3];
  const float* bk = (const float*)d_in[4];
  const float* Wv = (const float*)d_in[5];
  const float* bv = (const float*)d_in[6];
  const float* Wp = (const float*)d_in[7];
  const float* bp = (const float*)d_in[8];

  float* y_out = (float*)d_out;
  float* presentK = y_out + (size_t)MROWS * CDIM;                  // present[0] = k
  float* presentV = presentK + (size_t)BATCH * NHEAD * SEQ * HDIM; // present[1] = v

  char* ws = (char*)d_ws;
  const size_t XE  = (size_t)MROWS * CDIM;
  const size_t WE  = (size_t)CDIM * CDIM;
  const size_t QE  = (size_t)BATCH * NHEAD * SEQ * HDIM;
  _Float16* x16  = (_Float16*)ws;  ws += XE * 2;
  _Float16* Wcat = (_Float16*)ws;  ws += 4 * WE * 2;   // q|k|v|p
  _Float16* q16  = (_Float16*)ws;  ws += QE * 2;
  _Float16* k16  = (_Float16*)ws;  ws += QE * 2;
  _Float16* vT16 = (_Float16*)ws;  ws += QE * 2;
  _Float16* y16  = (_Float16*)ws;  ws += XE * 2;

  cvt_f32_to_f16<<<(int)((XE + 255) / 256), 256, 0, stream>>>(x, x16, (int)XE);
  cvt_f32_to_f16<<<(int)((WE + 255) / 256), 256, 0, stream>>>(Wq, Wcat + 0 * WE, (int)WE);
  cvt_f32_to_f16<<<(int)((WE + 255) / 256), 256, 0, stream>>>(Wk, Wcat + 1 * WE, (int)WE);
  cvt_f32_to_f16<<<(int)((WE + 255) / 256), 256, 0, stream>>>(Wv, Wcat + 2 * WE, (int)WE);
  cvt_f32_to_f16<<<(int)((WE + 255) / 256), 256, 0, stream>>>(Wp, Wcat + 3 * WE, (int)WE);

  qkv_gemm_kernel<<<dim3(N3 / 128, MROWS / 128), 256, 0, stream>>>(
      x16, Wcat, bq, bk, bv, q16, k16, vT16, presentK, presentV);

  attn_kernel<<<dim3(SEQ / 128, BATCH * NHEAD), 256, 0, stream>>>(
      q16, k16, vT16, y16);

  proj_gemm_kernel<<<dim3(CDIM / 128, MROWS / 128), 256, 0, stream>>>(
      y16, Wcat + 3 * WE, bp, y_out);
}
